// MultiHeadAttention4Dtransformer_75625784148174
// MI455X (gfx1250) — matrix operand order repaired and checked by reading, not compiled, not run on hardware
//
#include <hip/hip_runtime.h>
#include <hip/hip_bf16.h>
#include <math.h>

// ---------------------------------------------------------------------------
// CDNA5 (gfx1250) wave32 WMMA helpers: V_WMMA_F32_16X16X32_BF16
// Fragment layouts per CDNA5 ISA 7.12.2.
// ---------------------------------------------------------------------------
typedef __attribute__((ext_vector_type(16))) __bf16 v16bf;
typedef __attribute__((ext_vector_type(8)))  float  v8f;

union FragBF {
    v16bf v;
    unsigned int d[8];
};

#if defined(__has_builtin)
#if __has_builtin(__builtin_amdgcn_cvt_pk_bf16_f32)
#define HAVE_CVT_PK_BF16 1
#endif
#endif

__device__ __forceinline__ unsigned short f2bf(float f) {
    unsigned int x = __float_as_uint(f);
    x += 0x7FFFu + ((x >> 16) & 1u);          // round-to-nearest-even
    return (unsigned short)(x >> 16);
}
__device__ __forceinline__ unsigned int pack2bf(float lo, float hi) {
#ifdef HAVE_CVT_PK_BF16
    typedef __attribute__((ext_vector_type(2))) __bf16 v2bf;
    union { v2bf v; unsigned int u; } r;
    r.v = __builtin_amdgcn_cvt_pk_bf16_f32(lo, hi);   // v_cvt_pk_bf16_f32
    return r.u;
#else
    return (unsigned int)f2bf(lo) | ((unsigned int)f2bf(hi) << 16);
#endif
}

// A-matrix 16x32 (MxK), element (m,k) at base[m*ld + k].
// Lane l (0..15): M=l, K 0..7 in VGPR0..3, K 16..23 in VGPR4..7.
// Lane l+16: same M, K offset +8.
__device__ __forceinline__ v16bf load_a_bf16(const unsigned short* base, int ld, int lane) {
    FragBF f;
    const unsigned short* row = base + (lane & 15) * ld;
    int koff = (lane >> 4) * 8;
#pragma unroll
    for (int v = 0; v < 8; ++v) {
        int kb = (v < 4) ? (2 * v) : (16 + 2 * (v - 4));
        int k = kb + koff;
        f.d[v] = *(const unsigned int*)(row + k);   // k even -> dword aligned
    }
    return f.v;
}

// B-matrix 32x16 (KxN), fragment element (k,n) sourced from mem[n*ld + k]
// (row-major [N][K]); the WMMA then computes D = A * Mem^T.
// Lane l (0..15): N=l, K 0..7 in VGPR0..3, K 16..23 in VGPR4..7.  Lane l+16: K offset +8.
__device__ __forceinline__ v16bf load_b_bf16(const unsigned short* base, int ld, int lane) {
    FragBF f;
    const unsigned short* row = base + (lane & 15) * ld;
    int koff = (lane >> 4) * 8;
#pragma unroll
    for (int v = 0; v < 8; ++v) {
        int k = ((v < 4) ? (2 * v) : (16 + 2 * (v - 4))) + koff;
        f.d[v] = *(const unsigned int*)(row + k);
    }
    return f.v;
}

__device__ __forceinline__ v8f wmma_bf16(v16bf a, v16bf b, v8f c) {
    return __builtin_amdgcn_wmma_f32_16x16x32_bf16(false, a, false, b,
                                                   (short)0, c, false, false);
}

// C/D f32 16x16: VGPR r -> M = r + (lane>=16 ? 8 : 0), N = lane & 15.

// ---------------------------------------------------------------------------
// Problem constants
// ---------------------------------------------------------------------------
#define Bq   8
#define Sq   1024
#define Dq   512
#define Hq   8
#define Dhq  64
#define Mrows (Bq * Sq)          // 8192

// ---------------------------------------------------------------------------
// Kernel 0: f32 -> bf16 pair conversion (inputs q/k/v and the weights)
// ---------------------------------------------------------------------------
__global__ __launch_bounds__(256)
void cvt_f32_bf16(const float* __restrict__ src, unsigned int* __restrict__ dst,
                  int npairs) {
    int i = blockIdx.x * 256 + threadIdx.x;
    if (i < npairs) {
        const float2 p = ((const float2*)src)[i];
        dst[i] = pack2bf(p.x, p.y);
    }
}

// ---------------------------------------------------------------------------
// Kernel 1: projection GEMM  out[m,e] = sum_d X[m,d] * W[e,d] + bias[e]
// Xbf: [8192,512] bf16 row-major (A), Wbf: [512,512] bf16 row-major (B, [N][K]).
// Each wave: one 16(M) x 64(N) block -> A fragment reused by 4 accumulators.
// mode 0: out bf16 at [((b*H+h)*S+s)*64 + dh]   (Q / K head-major)
// mode 1: out bf16 at [((b*H+h)*64+dh)*S + s]   (V transposed)
// grid.x = (8192/16) * (512/64) = 4096 waves.
// ---------------------------------------------------------------------------
__global__ __launch_bounds__(32)
void proj_gemm_bf16out(const unsigned short* __restrict__ Xbf,
                       const unsigned short* __restrict__ Wbf,
                       const float* __restrict__ bias,
                       unsigned short* __restrict__ out, int mode) {
    int wid  = blockIdx.x;
    int mt   = wid >> 3;          // 512 M-tiles
    int ng   = wid & 7;           // 8 N-groups of 64 columns
    int lane = threadIdx.x;

    const unsigned short* Abase = Xbf + (size_t)(mt * 16) * Dq;
    const unsigned short* Bbase = Wbf + (size_t)(ng * 64) * Dq;

    v8f acc0 = {}, acc1 = {}, acc2 = {}, acc3 = {};
    for (int kk = 0; kk < Dq; kk += 32) {
        v16bf a = load_a_bf16(Abase + kk, Dq, lane);
        acc0 = wmma_bf16(a, load_b_bf16(Bbase + (size_t)(0 * 16) * Dq + kk, Dq, lane), acc0);
        acc1 = wmma_bf16(a, load_b_bf16(Bbase + (size_t)(1 * 16) * Dq + kk, Dq, lane), acc1);
        acc2 = wmma_bf16(a, load_b_bf16(Bbase + (size_t)(2 * 16) * Dq + kk, Dq, lane), acc2);
        acc3 = wmma_bf16(a, load_b_bf16(Bbase + (size_t)(3 * 16) * Dq + kk, Dq, lane), acc3);
    }

    int hi = lane >> 4;
    v8f accs[4] = {acc0, acc1, acc2, acc3};
#pragma unroll
    for (int t = 0; t < 4; ++t) {
        int n0 = ng * 64 + t * 16 + (lane & 15);
        float bn = bias[n0];
        int h  = n0 >> 6;
        int dh = n0 & 63;
#pragma unroll
        for (int r = 0; r < 8; ++r) {
            int mg = mt * 16 + r + hi * 8;
            int bb = mg >> 10;
            int ss = mg & 1023;
            float val = accs[t][r] + bn;
            size_t idx = (mode == 0)
                ? ((size_t)((bb * Hq + h) * Sq + ss) * Dhq + dh)
                : ((size_t)((bb * Hq + h) * Dhq + dh) * Sq + ss);
            out[idx] = f2bf(val);
        }
    }
}

// ---------------------------------------------------------------------------
// Kernel 2: fused attention for one (b, h, 16-query-row) stripe.
// QK^T (WMMA) -> LDS -> softmax -> tail-mass distance decay -> softmax ->
// scores to global + bf16 P in LDS -> PV (WMMA) -> bf16 attn-out.
// One wave per workgroup. grid.x = B*H*(S/16) = 4096.  LDS: 96 KB.
// ---------------------------------------------------------------------------
__global__ __launch_bounds__(32)
void attn_kernel(const unsigned short* __restrict__ Qbf,
                 const unsigned short* __restrict__ Kbf,
                 const unsigned short* __restrict__ VTbf,
                 const float* __restrict__ gammas,
                 float* __restrict__ scores_out,
                 unsigned short* __restrict__ Obf) {
    __shared__ float        sm[16 * Sq];        // 64 KB raw/decayed scores
    __shared__ unsigned int pbf[16 * Sq / 2];   // 32 KB packed-bf16 probs

    int wid  = blockIdx.x;
    int qt   = wid & 63;
    int h    = (wid >> 6) & 7;
    int b    = wid >> 9;
    int lane = threadIdx.x;
    int lane16 = lane & 15;
    int hi     = lane >> 4;
    int qbase  = qt * 16;

    const unsigned short* Qh = Qbf  + (size_t)(b * Hq + h) * Sq * Dhq;
    const unsigned short* Kh = Kbf  + (size_t)(b * Hq + h) * Sq * Dhq;
    const unsigned short* VT = VTbf + (size_t)(b * Hq + h) * Dhq * Sq;

    // Q fragments for this stripe (Dh=64 -> two K=32 fragments), loaded once.
    v16bf aq0 = load_a_bf16(Qh + (size_t)qbase * Dhq + 0,  Dhq, lane);
    v16bf aq1 = load_a_bf16(Qh + (size_t)qbase * Dhq + 32, Dhq, lane);

    // ---- Phase 1: raw scores (causal tiles only) into LDS ------------------
    for (int kt = 0; kt <= qt; ++kt) {
        if (kt < qt)   // prefetch next K tile (global_prefetch_b8)
            __builtin_prefetch(Kh + (size_t)((kt + 1) * 16) * Dhq, 0, 0);
        v16bf b0 = load_b_bf16(Kh + (size_t)(kt * 16) * Dhq + 0,  Dhq, lane);
        v16bf b1 = load_b_bf16(Kh + (size_t)(kt * 16) * Dhq + 32, Dhq, lane);
        v8f acc = {};
        acc = wmma_bf16(aq0, b0, acc);
        acc = wmma_bf16(aq1, b1, acc);
        int j = kt * 16 + lane16;
#pragma unroll
        for (int r = 0; r < 8; ++r) {
            int m = r + hi * 8;
            sm[(m << 10) + j] = acc[r] * 0.125f;     // 1/sqrt(64)
        }
    }

    float gamma = -fabsf(gammas[h]);

    // ---- Phase 2: per-row softmax / distance decay / second softmax --------
    // Chunked layout: lane owns columns [lane*32, lane*32+32).
    for (int m = 0; m < 16; ++m) {
        int qpos   = qbase + m;
        int nvalid = qpos + 1;                 // unmasked keys (causal)
        float* srow = sm + (m << 10);
        int j0 = lane << 5;
        float te[32];

        // max over valid
        float mx = -3.0e38f;
#pragma unroll
        for (int t = 0; t < 32; ++t) {
            int j = j0 + t;
            if (j < nvalid) mx = fmaxf(mx, srow[j]);
        }
        for (int off = 16; off; off >>= 1) mx = fmaxf(mx, __shfl_xor(mx, off, 32));

        // exp + per-lane chunk sum
        float ls = 0.f;
#pragma unroll
        for (int t = 0; t < 32; ++t) {
            int j = j0 + t;
            float e = (j < nvalid) ? __expf(srow[j] - mx) : 0.f;
            te[t] = e;
            ls += e;
        }
        float z = ls;
        for (int off = 16; off; off >>= 1) z += __shfl_xor(z, off, 32);
        float invz = 1.0f / z;

        // exclusive wave scan of chunk sums
        float sc = ls;
        for (int off = 1; off < 32; off <<= 1) {
            float nb = __shfl_up(sc, off, 32);
            if (lane >= off) sc += nb;
        }
        float running = sc - ls;               // exclusive prefix

        // tail-mass decay: disttotal = 1 (softmax), tail = 1 - cumsum/Z
        float mx2 = -3.0e38f;
#pragma unroll
        for (int t = 0; t < 32; ++t) {
            int j = j0 + t;
            if (j < nvalid) {
                running += te[t];
                float tail = fmaxf(1.0f - running * invz, 0.0f);
                float dist = sqrtf(tail * (float)(qpos - j));
                float eff  = fminf(fmaxf(__expf(gamma * dist), 1e-5f), 1e5f);
                float ns = srow[j] * eff;
                srow[j] = ns;
                mx2 = fmaxf(mx2, ns);
            }
        }
        for (int off = 16; off; off >>= 1) mx2 = fmaxf(mx2, __shfl_xor(mx2, off, 32));

        float z2 = 0.f;
#pragma unroll
        for (int t = 0; t < 32; ++t) {
            int j = j0 + t;
            float e = (j < nvalid) ? __expf(srow[j] - mx2) : 0.f;
            te[t] = e;
            z2 += e;
        }
        for (int off = 16; off; off >>= 1) z2 += __shfl_xor(z2, off, 32);
        float invz2 = 1.0f / z2;

        // final probs: global scores (hard zero on mask) + packed bf16 P (LDS)
        float* orow = scores_out + ((size_t)((b * Hq + h) * Sq + qpos) << 10);
        unsigned int* prow = pbf + (((m << 10) + j0) >> 1);
#pragma unroll
        for (int t = 0; t < 32; t += 2) {
            int j = j0 + t;
            float p0 = (j     < nvalid) ? te[t]     * invz2 : 0.0f;
            float p1 = (j + 1 < nvalid) ? te[t + 1] * invz2 : 0.0f;
            *(float2*)(orow + j) = float2{p0, p1};
            prow[t >> 1] = pack2bf(p0, p1);
        }
    }

    // ---- Phase 3: PV via WMMA; one P fragment feeds 4 Dh column tiles ------
    int kend = ((qt + 1) * 16 + 31) & ~31;     // round valid keys up to 32
    const unsigned short* pb = (const unsigned short*)pbf;
    v8f o0 = {}, o1 = {}, o2 = {}, o3 = {};
    for (int kk = 0; kk < kend; kk += 32) {
        v16bf ap = load_a_bf16(pb + kk, Sq, lane);                   // P (LDS bf16)
        o0 = wmma_bf16(ap, load_b_bf16(VT + (size_t)(0 * 16) * Sq + kk, Sq, lane), o0);
        o1 = wmma_bf16(ap, load_b_bf16(VT + (size_t)(1 * 16) * Sq + kk, Sq, lane), o1);
        o2 = wmma_bf16(ap, load_b_bf16(VT + (size_t)(2 * 16) * Sq + kk, Sq, lane), o2);
        o3 = wmma_bf16(ap, load_b_bf16(VT + (size_t)(3 * 16) * Sq + kk, Sq, lane), o3);
    }
    v8f os[4] = {o0, o1, o2, o3};
#pragma unroll
    for (int nt = 0; nt < 4; ++nt) {
#pragma unroll
        for (int r = 0; r < 8; ++r) {
            int sq2 = qbase + r + hi * 8;
            int dh  = nt * 16 + lane16;
            Obf[(size_t)(b * Sq + sq2) * Dq + h * Dhq + dh] = f2bf(os[nt][r]);
        }
    }
}

// ---------------------------------------------------------------------------
// Kernel 3: output projection  out[m,e] = sum_d Obf[m,d] * Wo[e,d] + bo[e]
// A and B both bf16; each wave computes 16(M) x 64(N).
// ---------------------------------------------------------------------------
__global__ __launch_bounds__(32)
void out_gemm(const unsigned short* __restrict__ A,
              const unsigned short* __restrict__ Wbf,
              const float* __restrict__ bias, float* __restrict__ out) {
    int wid  = blockIdx.x;
    int mt   = wid >> 3;
    int ng   = wid & 7;
    int lane = threadIdx.x;

    const unsigned short* Abase = A   + (size_t)(mt * 16) * Dq;
    const unsigned short* Bbase = Wbf + (size_t)(ng * 64) * Dq;

    v8f acc0 = {}, acc1 = {}, acc2 = {}, acc3 = {};
    for (int kk = 0; kk < Dq; kk += 32) {
        v16bf a = load_a_bf16(Abase + kk, Dq, lane);
        acc0 = wmma_bf16(a, load_b_bf16(Bbase + (size_t)(0 * 16) * Dq + kk, Dq, lane), acc0);
        acc1 = wmma_bf16(a, load_b_bf16(Bbase + (size_t)(1 * 16) * Dq + kk, Dq, lane), acc1);
        acc2 = wmma_bf16(a, load_b_bf16(Bbase + (size_t)(2 * 16) * Dq + kk, Dq, lane), acc2);
        acc3 = wmma_bf16(a, load_b_bf16(Bbase + (size_t)(3 * 16) * Dq + kk, Dq, lane), acc3);
    }

    int hi = lane >> 4;
    v8f accs[4] = {acc0, acc1, acc2, acc3};
#pragma unroll
    for (int t = 0; t < 4; ++t) {
        int n0 = ng * 64 + t * 16 + (lane & 15);
        float bn = bias[n0];
#pragma unroll
        for (int r = 0; r < 8; ++r) {
            int mg = mt * 16 + r + hi * 8;
            out[(size_t)mg * Dq + n0] = accs[t][r] + bn;
        }
    }
}

// ---------------------------------------------------------------------------
// Host launcher
// ---------------------------------------------------------------------------
extern "C" void kernel_launch(void* const* d_in, const int* in_sizes, int n_in,
                              void* d_out, int out_size, void* d_ws, size_t ws_size,
                              hipStream_t stream) {
    const float* q      = (const float*)d_in[0];
    const float* k      = (const float*)d_in[1];
    const float* v      = (const float*)d_in[2];
    // d_in[3] = causal mask (implicit in kernel; unused)
    const float* Wq     = (const float*)d_in[4];
    const float* bq     = (const float*)d_in[5];
    const float* Wv     = (const float*)d_in[6];
    const float* bv     = (const float*)d_in[7];
    const float* Wo     = (const float*)d_in[8];
    const float* bo     = (const float*)d_in[9];
    const float* gammas = (const float*)d_in[10];

    float* out        = (float*)d_out;                       // [B,S,D]
    float* scores_out = out + (size_t)Bq * Sq * Dq;          // [B,H,S,S]

    // Workspace layout (bf16 = unsigned short elements):
    //   Qbf/Kbf/VTbf/Obf : B*S*D each (8 MB each)
    //   Xq/Xk/Xv bf16    : B*S*D each (8 MB each)
    //   Wqbf/Wvbf/Wobf   : D*D each   (0.5 MB each)   total ~57.5 MB
    const size_t NE = (size_t)Bq * Sq * Dq;
    const size_t NW = (size_t)Dq * Dq;
    unsigned short* Qbf  = (unsigned short*)d_ws;
    unsigned short* Kbf  = Qbf  + NE;
    unsigned short* VTbf = Kbf  + NE;
    unsigned short* Obf  = VTbf + NE;
    unsigned short* Xqbf = Obf  + NE;
    unsigned short* Xkbf = Xqbf + NE;
    unsigned short* Xvbf = Xkbf + NE;
    unsigned short* Wqbf = Xvbf + NE;
    unsigned short* Wvbf = Wqbf + NW;
    unsigned short* Wobf = Wvbf + NW;

    dim3 blk32(32);

    // f32 -> bf16 conversions (each element exactly once)
    {
        int npw = (int)(NW / 2);               // 131072 pairs per weight
        dim3 gw((npw + 255) / 256), b256(256);
        cvt_f32_bf16<<<gw, b256, 0, stream>>>(Wq, (unsigned int*)Wqbf, npw);
        cvt_f32_bf16<<<gw, b256, 0, stream>>>(Wv, (unsigned int*)Wvbf, npw);
        cvt_f32_bf16<<<gw, b256, 0, stream>>>(Wo, (unsigned int*)Wobf, npw);
        int npx = (int)(NE / 2);               // 2M pairs per input
        dim3 gx((npx + 255) / 256);
        cvt_f32_bf16<<<gx, b256, 0, stream>>>(q, (unsigned int*)Xqbf, npx);
        cvt_f32_bf16<<<gx, b256, 0, stream>>>(k, (unsigned int*)Xkbf, npx);
        cvt_f32_bf16<<<gx, b256, 0, stream>>>(v, (unsigned int*)Xvbf, npx);
    }

    dim3 gproj((Mrows / 16) * (Dq / 64));      // 512 * 8 = 4096 waves
    proj_gemm_bf16out<<<gproj, blk32, 0, stream>>>(Xqbf, Wqbf, bq, Qbf, 0);
    proj_gemm_bf16out<<<gproj, blk32, 0, stream>>>(Xkbf, Wqbf, bq, Kbf, 0);
    proj_gemm_bf16out<<<gproj, blk32, 0, stream>>>(Xvbf, Wvbf, bv, VTbf, 1);

    dim3 gattn(Bq * Hq * (Sq / 16));           // 4096 workgroups, 1 wave each
    attn_kernel<<<gattn, blk32, 0, stream>>>(Qbf, Kbf, VTbf, gammas, scores_out, Obf);

    out_gemm<<<gproj, blk32, 0, stream>>>(Obf, Wobf, bo, out);
}
